// GATEncoder_74912819577120
// MI455X (gfx1250) — matrix operand order repaired and checked by reading, not compiled, not run on hardware
//
#include <hip/hip_runtime.h>
#include <cstdint>

// ---------------------------------------------------------------------------
// GAT encoder for MI455X (gfx1250, wave32, WMMA).
// GEMMs + output heads run through v_wmma_f32_16x16x32_f16 with vectorized
// (b128) fragment loads; LDS staging uses gfx1250 async global->LDS copies
// (ASYNCcnt) when the toolchain exposes them; edge phase is bandwidth-tuned
// scalar/atomic code.
// ---------------------------------------------------------------------------

typedef _Float16 v16h __attribute__((ext_vector_type(16)));
typedef _Float16 v8h  __attribute__((ext_vector_type(8)));
typedef float    v8f  __attribute__((ext_vector_type(8)));
typedef int      v4i  __attribute__((ext_vector_type(4)));

#define H_DIM 128
#define OUT_DIM 32
#define LRELU(v) ((v) > 0.0f ? (v) : 0.2f * (v))

#if defined(__has_builtin)
#if __has_builtin(__builtin_amdgcn_global_load_async_to_lds_b128) && \
    __has_builtin(__builtin_amdgcn_s_wait_asynccnt)
#define USE_ASYNC_LDS 1
#endif
#endif

#ifdef USE_ASYNC_LDS
// builtin signature (from hipcc diagnostic): arg0 = v4i addrspace(1)* (global
// src, printed as "__device__"), arg1 = v4i addrspace(3)* (LDS dst), then imm
// offset, imm cpol. generic->AS1: same 64-bit value; generic->AS3: LDS offset
// is addr[31:0] (ISA 10.2).
typedef __attribute__((address_space(1))) v4i as1_v4i;
typedef __attribute__((address_space(3))) v4i as3_v4i;
__device__ __forceinline__ void async_copy_b128(const void* g, void* l) {
    __builtin_amdgcn_global_load_async_to_lds_b128(
        (as1_v4i*)(uintptr_t)g,
        (as3_v4i*)(unsigned)(uintptr_t)l,
        0, 0);
}
#endif

__device__ __forceinline__ v16h cat16(v8h lo, v8h hi) {
    return __builtin_shufflevector(lo, hi, 0, 1, 2, 3, 4, 5, 6, 7,
                                           8, 9, 10, 11, 12, 13, 14, 15);
}

// ---------------- f32 -> f16 transposed weight conversion -------------------
// src is [K x C] row-major, dst is [C x K] row-major (i.e. src transposed)
__global__ void k_convert_t(const float* __restrict__ src, _Float16* __restrict__ dst,
                            int K, int C) {
    int i = blockIdx.x * blockDim.x + threadIdx.x;
    if (i < K * C) {
        int k = i / C, c = i % C;
        dst[c * K + k] = (_Float16)src[i];
    }
}

// ---------------- library-size + log1p normalize ----------------------------
__global__ void k_normalize(const float* __restrict__ x, _Float16* __restrict__ xh,
                            float* __restrict__ l_out, int f_in) {
    __shared__ float red[256];
    const int row = blockIdx.x;
    const int tid = threadIdx.x;
    const float* xr = x + (long)row * f_in;
    float s = 0.0f;
    for (int i = tid; i < f_in; i += 256) s += xr[i];
    red[tid] = s;
    __syncthreads();
    for (int off = 128; off > 0; off >>= 1) {
        if (tid < off) red[tid] += red[tid + off];
        __syncthreads();
    }
    const float l = red[0];
    if (tid == 0) l_out[row] = l;
    const float scale = 10000.0f / l;
    _Float16* xo = xh + (long)row * f_in;
    for (int i = tid; i < f_in; i += 256) xo[i] = (_Float16)log1pf(xr[i] * scale);
}

// ---------------- WMMA GEMM: C[N x 128] = A[N x K] * W[K x 128] -------------
// WT is W transposed: [128 x K] f16. Block = 256 thr = 8 waves, covers a
// 64-row x 128-col tile. Wave w owns cols [16w,16w+16) and 4 row sub-tiles
// -> 4 WMMAs per k-step sharing one B fragment.
#define APAD 40   // padded LDS row stride in halves (80B: 16B-aligned, bank-spread)
__global__ void __launch_bounds__(256)
k_gemm_wmma(const _Float16* __restrict__ A, const _Float16* __restrict__ WT,
            float* __restrict__ C, int K, int nrows) {
    __shared__ _Float16 As[64 * APAD];    // 64 rows x 32 k (padded)
    __shared__ _Float16 Bs[H_DIM * APAD]; // 128 cols x 32 k (padded, col-major slab)
    const int tid  = threadIdx.x;
    const int wave = tid >> 5;
    const int lane = tid & 31;
    const int row0 = blockIdx.x * 64;
    const int col0 = wave * 16;
    const int m    = lane & 15;
    const int kh   = (lane >> 4) * 8;     // unused (A and B both select k-half via koff)
    const int koff = (lane >> 4) * 16;    // B: k-half select

    v8f acc[4] = {};
    for (int k0 = 0; k0 < K; k0 += 32) {
#ifdef USE_ASYNC_LDS
        // async DMA staging: bypasses VGPRs, tracked by ASYNCcnt
        {   // A tile: 64 rows x 4 b128 segs = 256 copies, 1 per thread
            int r  = tid >> 2;
            int sg = (tid & 3) * 8;       // halves
            int rr = row0 + r; if (rr >= nrows) rr = nrows - 1;
            async_copy_b128(A + rr * K + k0 + sg, &As[r * APAD + sg]);
        }
        #pragma unroll
        for (int j = 0; j < 2; ++j) {     // B slab: 128 cols x 4 segs = 512 copies
            int i  = tid + j * 256;
            int c  = i >> 2;
            int sg = (i & 3) * 8;
            async_copy_b128(WT + c * K + k0 + sg, &Bs[c * APAD + sg]);
        }
        __builtin_amdgcn_s_wait_asynccnt(0);
#else
        // sync staging fallback: global -> VGPR -> LDS dwords
        #pragma unroll
        for (int j = 0; j < 4; ++j) {
            int i  = tid + j * 256;
            int r  = i >> 4;
            int kd = (i & 15) * 2;
            int rr = row0 + r; if (rr >= nrows) rr = nrows - 1;
            *(uint32_t*)&As[r * APAD + kd] =
                *(const uint32_t*)(A + rr * K + k0 + kd);
        }
        #pragma unroll
        for (int j = 0; j < 8; ++j) {
            int i  = tid + j * 256;
            int c  = i >> 4;
            int kd = (i & 15) * 2;
            *(uint32_t*)&Bs[c * APAD + kd] =
                *(const uint32_t*)(WT + c * K + k0 + kd);
        }
#endif
        __syncthreads();

        // B fragment: 16 contiguous halves of Bs[col0+n] -> two b128 loads
        const _Float16* br = &Bs[(col0 + (lane & 15)) * APAD + koff];
        v16h b = cat16(*(const v8h*)br, *(const v8h*)(br + 8));
        #pragma unroll
        for (int t = 0; t < 4; ++t) {
            // A fragment: 16 contiguous halves [koff,koff+16) -> two b128 loads
            const _Float16* ar = &As[(t * 16 + m) * APAD + koff];
            v16h a = cat16(*(const v8h*)ar, *(const v8h*)(ar + 8));
            acc[t] = __builtin_amdgcn_wmma_f32_16x16x32_f16(
                         false, a, false, b, (short)0, acc[t], false, false);
        }
        __syncthreads();
    }
    // D layout: VGPR r -> M = r (lanes 0-15) / r+8 (lanes 16-31), N = lane&15
    const int ncol = col0 + (lane & 15);
    const int mo   = (lane >> 4) * 8;
    if (row0 + 64 <= nrows) {
        #pragma unroll
        for (int t = 0; t < 4; ++t)
            #pragma unroll
            for (int r = 0; r < 8; ++r)
                C[(row0 + t * 16 + mo + r) * H_DIM + ncol] = acc[t][r];
    } else {
        #pragma unroll
        for (int t = 0; t < 4; ++t)
            #pragma unroll
            for (int r = 0; r < 8; ++r) {
                int row = row0 + t * 16 + mo + r;
                if (row < nrows) C[row * H_DIM + ncol] = acc[t][r];
            }
    }
}

// ---------------- alpha_s / alpha_d GEMV (wave per node) --------------------
__global__ void k_alpha(const float* __restrict__ h2, const float* __restrict__ a_s,
                        const float* __restrict__ a_d, float* __restrict__ as_out,
                        float* __restrict__ ad_out, int n_nodes) {
    const int gt   = blockIdx.x * blockDim.x + threadIdx.x;
    const int node = gt >> 5;
    const int lane = gt & 31;
    if (node >= n_nodes) return;
    const float* hr = h2 + (long)node * H_DIM;
    float s = 0.0f, d = 0.0f;
    #pragma unroll
    for (int j = 0; j < 4; ++j) {
        int f = lane + 32 * j;
        float v = hr[f];
        s += v * a_s[f];
        d += v * a_d[f];
    }
    #pragma unroll
    for (int off = 16; off > 0; off >>= 1) {
        s += __shfl_xor(s, off, 32);
        d += __shfl_xor(d, off, 32);
    }
    if (lane == 0) { as_out[node] = s; ad_out[node] = d; }
}

// ---------------- per-layer init (must run every call) ----------------------
__global__ void k_init(float* __restrict__ aggr, unsigned int* __restrict__ m_bits,
                       float* __restrict__ denom, float* __restrict__ bnstats,
                       long total, int n_nodes) {
    long i = (long)blockIdx.x * blockDim.x + threadIdx.x;
    long stride = (long)gridDim.x * blockDim.x;
    for (; i < total; i += stride) {
        aggr[i] = 0.0f;
        if (i < n_nodes) { m_bits[i] = 0xFF800000u; denom[i] = 0.0f; }  // -inf
        if (i < 512) bnstats[i] = 0.0f;
    }
}

// ---------------- edge pass 1: segment max of attention logits --------------
__global__ void k_edge_max(const int* __restrict__ src, const int* __restrict__ dst,
                           const float* __restrict__ as, const float* __restrict__ ad,
                           float* __restrict__ mbuf, long n_edges) {
    long e = (long)blockIdx.x * blockDim.x + threadIdx.x;
    if (e >= n_edges) return;
    int s = src[e], d = dst[e];
    float ev = as[s] + ad[d];
    ev = LRELU(ev);
    // ordered float atomic max: int-max for >=0, uint-min for <0
    if (ev >= 0.0f) atomicMax((int*)(mbuf + d), __float_as_int(ev));
    else            atomicMin((unsigned int*)(mbuf + d), __float_as_uint(ev));
}

// ---------------- edge pass 2: softmax-weighted aggregation -----------------
// 32 lanes per edge: lane j handles features j, j+32, j+64, j+96 (coalesced)
__global__ void k_edge_aggr(const int* __restrict__ src, const int* __restrict__ dst,
                            const float* __restrict__ as, const float* __restrict__ ad,
                            const float* __restrict__ mbuf, const float* __restrict__ h2,
                            float* __restrict__ denom, float* __restrict__ aggr,
                            long n_edges) {
    long gt = (long)blockIdx.x * blockDim.x + threadIdx.x;
    long e  = gt >> 5;
    int lane = (int)(gt & 31);
    if (e >= n_edges) return;
    int s = src[e], d = dst[e];
    float ev = as[s] + ad[d];
    ev = LRELU(ev);
    float w = __expf(ev - mbuf[d]);
    if (lane == 0) atomicAdd(denom + d, w);
    const float* hs = h2 + (long)s * H_DIM;
    float* ag = aggr + (long)d * H_DIM;
    #pragma unroll
    for (int j = 0; j < 4; ++j) {
        int f = lane + 32 * j;
        atomicAdd(ag + f, w * hs[f]);
    }
}

// ---------------- normalize by denom + BN statistics ------------------------
__global__ void k_node_norm(const float* __restrict__ aggr, const float* __restrict__ denom,
                            float* __restrict__ out, float* __restrict__ bnstats,
                            int n_nodes) {
    __shared__ float ls[256], ls2[256];
    const int tid  = threadIdx.x;
    const int f    = tid & 127;
    const int half = tid >> 7;
    float s = 0.0f, s2 = 0.0f;
    for (long n = (long)blockIdx.x * 2 + half; n < n_nodes; n += (long)gridDim.x * 2) {
        float v = aggr[n * H_DIM + f] / (denom[n] + 1e-16f);
        out[n * H_DIM + f] = v;
        s += v; s2 += v * v;
    }
    ls[tid] = s; ls2[tid] = s2;
    __syncthreads();
    if (tid < 128) {
        atomicAdd(bnstats + f,       ls[tid] + ls[tid + 128]);        // sum
        atomicAdd(bnstats + 128 + f, ls2[tid] + ls2[tid + 128]);      // sumsq
    }
}

// ---------------- BN finalize: scale/shift per feature ----------------------
__global__ void k_bn_finalize(float* __restrict__ bnstats, const float* __restrict__ g,
                              const float* __restrict__ b, float inv_n) {
    int f = threadIdx.x;
    if (f < H_DIM) {
        float mu  = bnstats[f] * inv_n;
        float var = bnstats[128 + f] * inv_n - mu * mu;
        float sc  = g[f] * rsqrtf(var + 1e-5f);
        bnstats[256 + f] = sc;              // scale
        bnstats[384 + f] = b[f] - mu * sc;  // shift
    }
}

// ---------------- BN apply + leaky relu + f16 copy --------------------------
__global__ void k_bn_apply(float* __restrict__ buf, const float* __restrict__ bnstats,
                           _Float16* __restrict__ buf_h16, long total) {
    long i = (long)blockIdx.x * blockDim.x + threadIdx.x;
    if (i >= total) return;
    int f = (int)(i & 127);
    float v = buf[i] * bnstats[256 + f] + bnstats[384 + f];
    v = LRELU(v);
    buf[i] = v;
    buf_h16[i] = (_Float16)v;
}

// ---------------- WMMA output heads: loc and softplus std -------------------
// Block = 256 thr = 8 waves, covers 32 rows. Wave w: rowHalf = w>>2,
// tile = w&3 (0,1 -> loc cols 0/16; 2,3 -> std cols 0/16). K = 128 unrolled.
__global__ void __launch_bounds__(256)
k_heads_wmma(const _Float16* __restrict__ ptrh, const _Float16* __restrict__ locWT,
             const _Float16* __restrict__ stdWT, const float* __restrict__ locb,
             const float* __restrict__ stdb, float* __restrict__ out, int n_nodes) {
    const int wave = threadIdx.x >> 5;
    const int lane = threadIdx.x & 31;
    const int tile = wave & 3;
    const int row0 = blockIdx.x * 32 + (wave >> 2) * 16;
    const int col0 = (tile & 1) * 16;
    const bool is_std = tile >= 2;
    const _Float16* WT = is_std ? stdWT : locWT;

    const int m    = lane & 15;
    const int kh   = (lane >> 4) * 8;
    const int koff = (lane >> 4) * 16;
    int arow = row0 + m; if (arow >= n_nodes) arow = n_nodes - 1;
    const _Float16* ar = ptrh + arow * H_DIM;
    const _Float16* br = WT + (col0 + (lane & 15)) * H_DIM;

    v8f acc = {};
    #pragma unroll
    for (int k0 = 0; k0 < H_DIM; k0 += 32) {
        v16h a = cat16(*(const v8h*)(ar + k0 + koff), *(const v8h*)(ar + k0 + koff + 8));
        v16h b = cat16(*(const v8h*)(br + k0 + koff), *(const v8h*)(br + k0 + koff + 8));
        acc = __builtin_amdgcn_wmma_f32_16x16x32_f16(
                  false, a, false, b, (short)0, acc, false, false);
    }
    const int ncol = col0 + (lane & 15);
    const int mo   = (lane >> 4) * 8;
    const float bias = (is_std ? stdb : locb)[ncol];
    const int base = is_std ? n_nodes * OUT_DIM : 0;
    #pragma unroll
    for (int r = 0; r < 8; ++r) {
        int row = row0 + mo + r;
        if (row < n_nodes) {
            float v = acc[r] + bias;
            if (is_std) {
                v = (v > 30.0f) ? v : log1pf(__expf(v));   // softplus
                v += 1e-7f;
            }
            out[base + row * OUT_DIM + ncol] = v;
        }
    }
}

// ---------------------------------------------------------------------------
extern "C" void kernel_launch(void* const* d_in, const int* in_sizes, int n_in,
                              void* d_out, int out_size, void* d_ws, size_t ws_size,
                              hipStream_t stream) {
    const float* x    = (const float*)d_in[0];
    const int*   ei   = (const int*)d_in[1];     // [2, E]: src row then dst row
    const float* W0   = (const float*)d_in[2];
    const float* a_s0 = (const float*)d_in[3];
    const float* a_d0 = (const float*)d_in[4];
    const float* g0   = (const float*)d_in[5];
    const float* b0   = (const float*)d_in[6];
    const float* W1   = (const float*)d_in[7];
    const float* a_s1 = (const float*)d_in[8];
    const float* a_d1 = (const float*)d_in[9];
    const float* g1   = (const float*)d_in[10];
    const float* b1   = (const float*)d_in[11];
    const float* locW = (const float*)d_in[12];
    const float* locb = (const float*)d_in[13];
    const float* stdW = (const float*)d_in[14];
    const float* stdb = (const float*)d_in[15];
    float* out = (float*)d_out;

    const int  F_IN = 1024;
    const int  N    = in_sizes[0] / F_IN;     // 100000
    const long E    = in_sizes[1] / 2;        // 1600000
    const int* src  = ei;
    const int* dst  = ei + E;

    // -------- workspace carve-out (256B aligned) --------
    auto align = [](size_t v) { return (v + 255) & ~(size_t)255; };
    char* ws = (char*)d_ws;
    size_t off = 0;
    _Float16* xh    = (_Float16*)(ws + off); off = align(off + (size_t)N * F_IN * 2);
    _Float16* w0t   = (_Float16*)(ws + off); off = align(off + (size_t)F_IN * H_DIM * 2);
    _Float16* w1t   = (_Float16*)(ws + off); off = align(off + (size_t)H_DIM * H_DIM * 2);
    _Float16* locWT = (_Float16*)(ws + off); off = align(off + (size_t)H_DIM * OUT_DIM * 2);
    _Float16* stdWT = (_Float16*)(ws + off); off = align(off + (size_t)H_DIM * OUT_DIM * 2);
    float*    h2    = (float*)(ws + off);    off = align(off + (size_t)N * H_DIM * 4);
    float*    ptrf  = (float*)(ws + off);    off = align(off + (size_t)N * H_DIM * 4);
    _Float16* ptrh  = (_Float16*)(ws + off); off = align(off + (size_t)N * H_DIM * 2);
    float*    aggr  = (float*)(ws + off);    off = align(off + (size_t)N * H_DIM * 4);
    float*    as_b  = (float*)(ws + off);    off = align(off + (size_t)N * 4);
    float*    ad_b  = (float*)(ws + off);    off = align(off + (size_t)N * 4);
    float*    mbuf  = (float*)(ws + off);    off = align(off + (size_t)N * 4);
    float*    denom = (float*)(ws + off);    off = align(off + (size_t)N * 4);
    float*    bnst  = (float*)(ws + off);    off = align(off + 512 * 4);
    (void)ws_size; (void)n_in; (void)out_size;

    const long NH        = (long)N * H_DIM;
    const int  row_tiles = (N + 63) / 64;
    const long edge_thr  = E * 32;
    const float inv_n    = 1.0f / (float)N;

    // transposed f16 weight conversions (small)
    k_convert_t<<<(F_IN * H_DIM + 255) / 256, 256, 0, stream>>>(W0, w0t, F_IN, H_DIM);
    k_convert_t<<<(H_DIM * H_DIM + 255) / 256, 256, 0, stream>>>(W1, w1t, H_DIM, H_DIM);
    k_convert_t<<<(H_DIM * OUT_DIM + 255) / 256, 256, 0, stream>>>(locW, locWT, H_DIM, OUT_DIM);
    k_convert_t<<<(H_DIM * OUT_DIM + 255) / 256, 256, 0, stream>>>(stdW, stdWT, H_DIM, OUT_DIM);

    // library size + log1p normalize -> f16 A matrix; l -> tail of output
    k_normalize<<<N, 256, 0, stream>>>(x, xh, out + 2L * N * OUT_DIM, F_IN);

    // ---------------- layer 0 ----------------
    k_gemm_wmma<<<row_tiles, 256, 0, stream>>>(xh, w0t, h2, F_IN, N);
    k_alpha<<<(N * 32 + 255) / 256, 256, 0, stream>>>(h2, a_s0, a_d0, as_b, ad_b, N);
    k_init<<<2048, 256, 0, stream>>>(aggr, (unsigned int*)mbuf, denom, bnst, NH, N);
    k_edge_max<<<(int)((E + 255) / 256), 256, 0, stream>>>(src, dst, as_b, ad_b, mbuf, E);
    k_edge_aggr<<<(int)((edge_thr + 255) / 256), 256, 0, stream>>>(
        src, dst, as_b, ad_b, mbuf, h2, denom, aggr, E);
    k_node_norm<<<512, 256, 0, stream>>>(aggr, denom, ptrf, bnst, N);
    k_bn_finalize<<<1, 128, 0, stream>>>(bnst, g0, b0, inv_n);
    k_bn_apply<<<(int)((NH + 255) / 256), 256, 0, stream>>>(ptrf, bnst, ptrh, NH);

    // ---------------- layer 1 ----------------
    k_gemm_wmma<<<row_tiles, 256, 0, stream>>>(ptrh, w1t, h2, H_DIM, N);
    k_alpha<<<(N * 32 + 255) / 256, 256, 0, stream>>>(h2, a_s1, a_d1, as_b, ad_b, N);
    k_init<<<2048, 256, 0, stream>>>(aggr, (unsigned int*)mbuf, denom, bnst, NH, N);
    k_edge_max<<<(int)((E + 255) / 256), 256, 0, stream>>>(src, dst, as_b, ad_b, mbuf, E);
    k_edge_aggr<<<(int)((edge_thr + 255) / 256), 256, 0, stream>>>(
        src, dst, as_b, ad_b, mbuf, h2, denom, aggr, E);
    k_node_norm<<<512, 256, 0, stream>>>(aggr, denom, ptrf, bnst, N);
    k_bn_finalize<<<1, 128, 0, stream>>>(bnst, g1, b1, inv_n);
    k_bn_apply<<<(int)((NH + 255) / 256), 256, 0, stream>>>(ptrf, bnst, ptrh, NH);

    // ---------------- WMMA heads ----------------
    k_heads_wmma<<<(N + 31) / 32, 256, 0, stream>>>(
        ptrh, locWT, stdWT, locb, stdb, out, N);
}
